// RelationAwareAttention_42288247996691
// MI455X (gfx1250) — matrix operand order repaired and checked by reading, not compiled, not run on hardware
//
#include <hip/hip_runtime.h>
#include <hip/hip_bf16.h>
#include <math.h>

typedef __bf16 bf16_t;
typedef __attribute__((ext_vector_type(16))) __bf16 v16bf;
typedef __attribute__((ext_vector_type(8)))  __bf16 v8bf;
typedef __attribute__((ext_vector_type(8)))  float  v8f;

#define N_SEQ 1024
#define DKDIM 128
#define DVDIM 128
#define KRELW 10
#define NREL  21

// ---------------- fragment load helpers ----------------
// A fragment (16x32 bf16): lane l holds row (l&15); K = g*8+0..7 and 16+g*8+0..7
__device__ __forceinline__ v16bf load_afrag(const bf16_t* __restrict__ aRow, int kc, int g) {
  v8bf lo = *(const v8bf*)(aRow + kc + g * 8);
  v8bf hi = *(const v8bf*)(aRow + kc + 16 + g * 8);
  v16bf a;
#pragma unroll
  for (int i = 0; i < 8; i++) { a[i] = lo[i]; a[i + 8] = hi[i]; }
  return a;
}
// B fragment (32x16 bf16): lane l holds column (l&15); K = g*8+0..7 and 16+g*8+0..7
__device__ __forceinline__ v16bf load_bfrag(const bf16_t* __restrict__ bRow, int kc, int g) {
  v8bf lo = *(const v8bf*)(bRow + kc + g * 8);
  v8bf hi = *(const v8bf*)(bRow + kc + 16 + g * 8);
  v16bf b;
#pragma unroll
  for (int i = 0; i < 8; i++) { b[i] = lo[i]; b[i + 8] = hi[i]; }
  return b;
}

// ---------------- conversion kernels ----------------

__global__ __launch_bounds__(256) void cvt_bf16_kernel(const float* __restrict__ in,
                                                       bf16_t* __restrict__ out, int n) {
  int i = blockIdx.x * 256 + threadIdx.x;
  if (i < n) out[i] = (bf16_t)in[i];
}

// in [rows, cols] row-major (f32)  ->  out [cols, rows] row-major (bf16)
__global__ __launch_bounds__(256) void cvt_transpose_kernel(const float* __restrict__ in,
                                                            bf16_t* __restrict__ out,
                                                            int rows, int cols) {
  int i = blockIdx.x * 256 + threadIdx.x;
  if (i < rows * cols) {
    int r = i / cols, c = i % cols;
    out[(size_t)c * rows + r] = (bf16_t)in[i];
  }
}

// ---------------- generic bf16 WMMA GEMM ----------------
// C[M,N] = scale * A[M,K] @ Bt[N,K]^T   (Bt stored N-major, i.e. B transposed)
// Each wave computes a 16x32 strip (two 16x16 tiles, two accumulators, A reused).
// Software-pipelined: next k-chunk fragments are loaded before current WMMAs.
// OUTMODE: 0 = f32 row-major, 1 = bf16 row-major, 2 = bf16 transposed ([N,M])
template<int OUTMODE>
__global__ __launch_bounds__(256) void gemm_wmma_kernel(const bf16_t* __restrict__ A,
                                                        const bf16_t* __restrict__ Bt,
                                                        void* __restrict__ Cv,
                                                        int M, int N, int Kd, float scale) {
  const int lane = threadIdx.x & 31;
  const int wave = threadIdx.x >> 5;
  const int g    = lane >> 4;
  const int ln   = lane & 15;
  const int tilesNW = N >> 5;                 // 32-wide n strips
  const int wt = blockIdx.x * 8 + wave;
  const int m0 = (wt / tilesNW) << 4;
  const int n0 = (wt % tilesNW) << 5;

  const bf16_t* aRow  = A  + (size_t)(m0 + ln) * Kd;
  const bf16_t* b0Row = Bt + (size_t)(n0 + ln) * Kd;
  const bf16_t* b1Row = Bt + (size_t)(n0 + 16 + ln) * Kd;

  v8f acc0 = {}, acc1 = {};
  v16bf a  = load_afrag(aRow, 0, g);
  v16bf b0 = load_bfrag(b0Row, 0, g);
  v16bf b1 = load_bfrag(b1Row, 0, g);

  for (int kc = 0; kc < Kd; kc += 32) {
    const int kn = (kc + 32 < Kd) ? kc + 32 : 0;   // branchless prefetch (last iter dummy)
    v16bf an  = load_afrag(aRow, kn, g);
    v16bf b0n = load_bfrag(b0Row, kn, g);
    v16bf b1n = load_bfrag(b1Row, kn, g);
    acc0 = __builtin_amdgcn_wmma_f32_16x16x32_bf16(false, a, false, b0,
                                                   (short)0, acc0, false, false);
    acc1 = __builtin_amdgcn_wmma_f32_16x16x32_bf16(false, a, false, b1,
                                                   (short)0, acc1, false, false);
    a = an; b0 = b0n; b1 = b1n;
  }

#pragma unroll
  for (int j = 0; j < 8; j++) {
    int m  = m0 + j + g * 8;       // D layout: VGPR j, hi half-wave -> M += 8
    int nA = n0 + ln;
    int nB = n0 + 16 + ln;
    float v0 = acc0[j] * scale;
    float v1 = acc1[j] * scale;
    if (OUTMODE == 0) {
      ((float*)Cv)[(size_t)m * N + nA] = v0;
      ((float*)Cv)[(size_t)m * N + nB] = v1;
    } else if (OUTMODE == 1) {
      ((bf16_t*)Cv)[(size_t)m * N + nA] = (bf16_t)v0;
      ((bf16_t*)Cv)[(size_t)m * N + nB] = (bf16_t)v1;
    } else {
      ((bf16_t*)Cv)[(size_t)nA * M + m] = (bf16_t)v0;
      ((bf16_t*)Cv)[(size_t)nB * M + m] = (bf16_t)v1;
    }
  }
}

// ---------------- relative-position bias table ----------------
// Bt[n][r] = Qs[n,:] . wk[r,:]   (1024 x 21, tiny)
__global__ __launch_bounds__(32) void bias_table_kernel(const bf16_t* __restrict__ Qs,
                                                        const float* __restrict__ wk,
                                                        float* __restrict__ Bt) {
  const int n = blockIdx.x;
  const int lane = threadIdx.x;
  float q[4];
#pragma unroll
  for (int i = 0; i < 4; i++) q[i] = (float)Qs[(size_t)n * DKDIM + lane + 32 * i];
  for (int r = 0; r < NREL; r++) {
    float s = 0.f;
#pragma unroll
    for (int i = 0; i < 4; i++) s += q[i] * wk[r * DKDIM + lane + 32 * i];
#pragma unroll
    for (int off = 16; off >= 1; off >>= 1) s += __shfl_xor(s, off);
    if (lane == 0) Bt[n * NREL + r] = s;
  }
}

// ---------------- fused S = Qs@K^T + rel-bias, softmax ----------------
// Block: 256 threads (8 waves), 16 rows per block. Qs fragments hoisted out of
// the column-tile loop; K fragments double-buffered. Writes f32 A and bf16 P.
#define SROW_STRIDE 1040   // 1024 + 16 floats: shifts banks by 16 -> conflict-free halves

__global__ __launch_bounds__(256) void attn_softmax_kernel(const bf16_t* __restrict__ Qs,
                                                           const bf16_t* __restrict__ Kb,
                                                           const float* __restrict__ Bt,
                                                           float* __restrict__ Aout,
                                                           bf16_t* __restrict__ Pout) {
  __shared__ float Srow[16 * SROW_STRIDE];
  __shared__ float BtS[16 * NREL];

  const int tid  = threadIdx.x;
  const int lane = tid & 31;
  const int wave = tid >> 5;
  const int g    = lane >> 4;
  const int ln   = lane & 15;
  const int m0   = blockIdx.x << 4;

  for (int i = tid; i < 16 * NREL; i += 256)
    BtS[i] = Bt[(size_t)(m0 + i / NREL) * NREL + (i % NREL)];

  // A fragments (rows m0..m0+15 of Qs) — invariant across all 8 column tiles.
  const bf16_t* aRow = Qs + (size_t)(m0 + ln) * DKDIM;
  v16bf afrag[4];
#pragma unroll
  for (int k4 = 0; k4 < 4; k4++) afrag[k4] = load_afrag(aRow, k4 * 32, g);

  __syncthreads();

  // Preload B fragments for the first tile.
  v16bf bcur[4];
  {
    const bf16_t* bRow = Kb + (size_t)((wave << 4) + ln) * DKDIM;
#pragma unroll
    for (int k4 = 0; k4 < 4; k4++) bcur[k4] = load_bfrag(bRow, k4 * 32, g);
  }

  for (int t = 0; t < 8; t++) {
    const int n0 = (wave + 8 * t) << 4;

    // Prefetch next tile's B fragments before the WMMAs (dummy reload on last t).
    const int tn = (t < 7) ? t + 1 : t;
    const bf16_t* bRowN = Kb + (size_t)(((wave + 8 * tn) << 4) + ln) * DKDIM;
    v16bf bnext[4];
#pragma unroll
    for (int k4 = 0; k4 < 4; k4++) bnext[k4] = load_bfrag(bRowN, k4 * 32, g);

    v8f acc = {};
#pragma unroll
    for (int k4 = 0; k4 < 4; k4++)
      acc = __builtin_amdgcn_wmma_f32_16x16x32_bf16(false, afrag[k4], false, bcur[k4],
                                                    (short)0, acc, false, false);

#pragma unroll
    for (int j = 0; j < 8; j++) {
      int row = j + g * 8;           // row within the 16-row block
      int col = n0 + ln;             // global column
      int rel = col - (m0 + row);
      rel = rel < -KRELW ? -KRELW : (rel > KRELW ? KRELW : rel);
      Srow[row * SROW_STRIDE + col] = acc[j] + BtS[row * NREL + rel + KRELW];
    }

#pragma unroll
    for (int k4 = 0; k4 < 4; k4++) bcur[k4] = bnext[k4];
  }
  __syncthreads();

  // Softmax: each half-wave (16 lanes) owns one row; row = wave*2 + g.
  const int row = wave * 2 + g;
  float* sp = Srow + row * SROW_STRIDE;

  float mx = -3.4e38f;
  for (int i = 0; i < 64; i++) mx = fmaxf(mx, sp[ln + 16 * i]);
#pragma unroll
  for (int off = 8; off >= 1; off >>= 1) mx = fmaxf(mx, __shfl_xor(mx, off));

  float sum = 0.f;
  for (int i = 0; i < 64; i++) {
    float e = __expf(sp[ln + 16 * i] - mx);
    sp[ln + 16 * i] = e;
    sum += e;
  }
#pragma unroll
  for (int off = 8; off >= 1; off >>= 1) sum += __shfl_xor(sum, off);
  const float inv = 1.f / sum;

  const size_t rb = (size_t)(m0 + row) * N_SEQ;
  for (int i = 0; i < 64; i++) {
    int c = ln + 16 * i;
    float p = sp[c] * inv;
    Aout[rb + c] = p;
    Pout[rb + c] = (bf16_t)p;
  }
}

// ---------------- launcher ----------------

extern "C" void kernel_launch(void* const* d_in, const int* in_sizes, int n_in,
                              void* d_out, int out_size, void* d_ws, size_t ws_size,
                              hipStream_t stream) {
  const float* Xma = (const float*)d_in[0];   // [1024,1024]
  const float* Xmi = (const float*)d_in[1];   // [1024,1024]
  const float* Wq  = (const float*)d_in[2];   // [1024,128]
  const float* Wk  = (const float*)d_in[3];   // [1024,128]
  const float* Wv  = (const float*)d_in[4];   // [1024,128]
  const float* Wo  = (const float*)d_in[5];   // [128,128]
  const float* wk  = (const float*)d_in[6];   // [21,128]

  float* Y    = (float*)d_out;                // [1024,128]
  float* Aout = Y + (size_t)N_SEQ * DVDIM;    // [1024,1024]

  char* w = (char*)d_ws;
  bf16_t* bXma = (bf16_t*)(w + 0);            // 1024*1024 bf16 = 2 MB
  bf16_t* bXmi = (bf16_t*)(w + 2097152);      // 2 MB
  bf16_t* bWqT = (bf16_t*)(w + 4194304);      // [128,1024] bf16 = 256 KB
  bf16_t* bWkT = (bf16_t*)(w + 4456448);
  bf16_t* bWvT = (bf16_t*)(w + 4718592);
  bf16_t* bWoT = (bf16_t*)(w + 4980736);      // [128,128]  bf16 = 32 KB
  bf16_t* bQs  = (bf16_t*)(w + 5013504);      // [1024,128] bf16 (scale folded)
  bf16_t* bKb  = (bf16_t*)(w + 5275648);      // [1024,128] bf16
  bf16_t* bVt  = (bf16_t*)(w + 5537792);      // [128,1024] bf16 (V transposed)
  float*  fBt  = (float*) (w + 5799936);      // [1024,21]  f32
  bf16_t* bP   = (bf16_t*)(w + 5885952);      // [1024,1024] bf16 probs
  bf16_t* bPV  = (bf16_t*)(w + 7983104);      // [1024,128] bf16

  const int nx = N_SEQ * N_SEQ;
  cvt_bf16_kernel<<<(nx + 255) / 256, 256, 0, stream>>>(Xma, bXma, nx);
  cvt_bf16_kernel<<<(nx + 255) / 256, 256, 0, stream>>>(Xmi, bXmi, nx);

  const int nw = 1024 * 128;
  cvt_transpose_kernel<<<(nw + 255) / 256, 256, 0, stream>>>(Wq, bWqT, 1024, 128);
  cvt_transpose_kernel<<<(nw + 255) / 256, 256, 0, stream>>>(Wk, bWkT, 1024, 128);
  cvt_transpose_kernel<<<(nw + 255) / 256, 256, 0, stream>>>(Wv, bWvT, 1024, 128);
  cvt_transpose_kernel<<<(128 * 128 + 255) / 256, 256, 0, stream>>>(Wo, bWoT, 128, 128);

  const float scale = 0.08838834764831845f;   // 1/sqrt(128)
  const int gQKV = (1024 / 16) * (128 / 32) / 8;   // 32 blocks (16x32 strip per wave)

  gemm_wmma_kernel<1><<<gQKV, 256, 0, stream>>>(bXma, bWqT, bQs, 1024, 128, 1024, scale);
  gemm_wmma_kernel<1><<<gQKV, 256, 0, stream>>>(bXmi, bWkT, bKb, 1024, 128, 1024, 1.f);
  gemm_wmma_kernel<2><<<gQKV, 256, 0, stream>>>(bXmi, bWvT, bVt, 1024, 128, 1024, 1.f);

  bias_table_kernel<<<1024, 32, 0, stream>>>(bQs, wk, fBt);

  attn_softmax_kernel<<<64, 256, 0, stream>>>(bQs, bKb, fBt, Aout, bP);

  gemm_wmma_kernel<1><<<gQKV, 256, 0, stream>>>(bP, bVt, bPV, 1024, 128, 1024, 1.f);
  gemm_wmma_kernel<0><<<gQKV, 256, 0, stream>>>(bPV, bWoT, Y, 1024, 128, 128, 1.f);
}
